// ResBlock_12240656793712
// MI455X (gfx1250) — matrix operand order repaired and checked by reading, not compiled, not run on hardware
//
#include <hip/hip_runtime.h>
#include <cstddef>
#include <cstdint>

// ---------------- problem constants ----------------
constexpr int B_  = 4;
constexpr int PER = 50000;
constexpr int N_  = B_ * PER;       // 200000 voxels
constexpr int CC  = 256;            // channels
constexpr int CE  = 1024;           // embedding dim
constexpr int G_  = 32;             // groups
constexpr int CG  = CC / G_;        // 8 channels per group
constexpr int KT  = 27;             // kernel taps
constexpr float EPS = 1e-5f;
constexpr int WK  = KT * CC * CC;   // weight elements per conv

// conv tiling
constexpr int ROWS = 64;            // voxel rows per block
constexpr int COB  = 64;            // output channels per block (grid.y = 4)
constexpr int SP   = 264;           // padded LDS row stride in bf16 (conflict-free b128)

typedef __attribute__((ext_vector_type(16))) __bf16 v16bf;
typedef __attribute__((ext_vector_type(8)))  float  v8f;

__device__ __forceinline__ float silu_f(float v) {
    return v / (1.0f + expf(-v));
}

// ---------------- CDNA5 async global->LDS copy (16B per lane) ----------------
__device__ __forceinline__ void async_cp16(void* lds, const void* g) {
    asm volatile("global_load_async_to_lds_b128 %0, %1, off"
                 :: "v"((unsigned)(uintptr_t)lds),
                    "v"((unsigned long long)(uintptr_t)g)
                 : "memory");
}
__device__ __forceinline__ void wait_async0() {
    asm volatile("s_wait_asynccnt 0x0" ::: "memory");
}

// ---------------- zero-init small stats region ----------------
__global__ void zero_kernel(float* __restrict__ p, int n) {
    int i = blockIdx.x * blockDim.x + threadIdx.x;
    if (i < n) p[i] = 0.0f;
}

// ---------------- weight split: fp32 [k][ci][co] -> bf16 hi/lo transposed [k][co][ci] ----------------
__global__ __launch_bounds__(256) void wsplit_kernel(
    const float* __restrict__ W, __bf16* __restrict__ hi, __bf16* __restrict__ lo)
{
    int gid = blockIdx.x * blockDim.x + threadIdx.x;
    if (gid >= WK) return;
    int k   = gid >> 16;
    int rem = gid & 65535;
    int ci  = rem >> 8;
    int co  = rem & 255;
    float w = W[gid];                                   // coalesced read
    size_t o = ((size_t)k << 16) | (size_t)(co << 8) | (size_t)ci;
    __bf16 h = (__bf16)w;
    hi[o] = h;
    lo[o] = (__bf16)(w - (float)h);
}

// ---------------- GroupNorm stats: per (grid b, group g) sum / sumsq / count ----------------
__global__ __launch_bounds__(256) void gn_stats_kernel(
    const float* __restrict__ x, const int* __restrict__ jidx,
    float* __restrict__ sums, float* __restrict__ sqs, float* __restrict__ cnts)
{
    __shared__ float ls[B_ * G_];
    __shared__ float lq[B_ * G_];
    __shared__ float lc[B_];
    const int tid = threadIdx.x;
    if (tid < B_ * G_) { ls[tid] = 0.0f; lq[tid] = 0.0f; }
    if (tid < B_) lc[tid] = 0.0f;
    __syncthreads();

    long long gid = (long long)blockIdx.x * blockDim.x + tid;   // one thread per (voxel, group)
    if (gid < (long long)N_ * G_) {
        int n = (int)(gid >> 5);
        int g = (int)(gid & 31);
        const float4* p = (const float4*)(x + (size_t)n * CC + g * CG);
        float4 a = p[0], b4 = p[1];
        float s = a.x + a.y + a.z + a.w + b4.x + b4.y + b4.z + b4.w;
        float q = a.x*a.x + a.y*a.y + a.z*a.z + a.w*a.w
                + b4.x*b4.x + b4.y*b4.y + b4.z*b4.z + b4.w*b4.w;
        int b = jidx[n];
        atomicAdd(&ls[b * G_ + g], s);
        atomicAdd(&lq[b * G_ + g], q);
        if (g == 0) atomicAdd(&lc[b], 1.0f);
    }
    __syncthreads();
    if (tid < B_ * G_) {
        atomicAdd(&sums[tid], ls[tid]);
        atomicAdd(&sqs[tid],  lq[tid]);
    }
    if (tid < B_) atomicAdd(&cnts[tid], lc[tid]);
}

// ---------------- GN apply + optional FiLM + SiLU -> split bf16 hi/lo activations ----------------
__global__ __launch_bounds__(256) void gn_apply_kernel(
    const float* __restrict__ x, const int* __restrict__ jidx,
    const float* __restrict__ sums, const float* __restrict__ sqs,
    const float* __restrict__ cnts,
    const float* __restrict__ gw, const float* __restrict__ gb,
    const float* __restrict__ film,       // nullptr or [B][512] = scale(256)||shift(256)
    __bf16* __restrict__ ohi, __bf16* __restrict__ olo)
{
    long long gid = (long long)blockIdx.x * blockDim.x + threadIdx.x;  // one thread per float4
    if (gid >= (long long)N_ * (CC / 4)) return;
    int n  = (int)(gid >> 6);
    int c  = ((int)(gid & 63)) * 4;
    int g  = c >> 3;
    int b  = jidx[n];
    float denom = cnts[b] * (float)CG;
    float mean  = sums[b * G_ + g] / denom;
    float var   = sqs [b * G_ + g] / denom - mean * mean;
    float rs    = rsqrtf(var + EPS);

    float4 xv = *(const float4*)(x  + (size_t)n * CC + c);
    float4 wv = *(const float4*)(gw + c);
    float4 bv = *(const float4*)(gb + c);
    float4 sc = make_float4(0.f, 0.f, 0.f, 0.f);
    float4 sh = sc;
    if (film) {
        sc = *(const float4*)(film + (size_t)b * 2 * CC + c);
        sh = *(const float4*)(film + (size_t)b * 2 * CC + CC + c);
    }
    float xi[4] = {xv.x, xv.y, xv.z, xv.w};
    float wi[4] = {wv.x, wv.y, wv.z, wv.w};
    float bi[4] = {bv.x, bv.y, bv.z, bv.w};
    float si[4] = {sc.x, sc.y, sc.z, sc.w};
    float hi[4] = {sh.x, sh.y, sh.z, sh.w};
    union { __bf16 b[4]; uint2 u; } ph, pl;
#pragma unroll
    for (int j = 0; j < 4; ++j) {
        float t = (xi[j] - mean) * rs * wi[j] + bi[j];
        t = t * (1.0f + si[j]) + hi[j];
        float o = silu_f(t);
        __bf16 hbv = (__bf16)o;
        ph.b[j] = hbv;
        pl.b[j] = (__bf16)(o - (float)hbv);
    }
    *(uint2*)(ohi + (size_t)n * CC + c) = ph.u;
    *(uint2*)(olo + (size_t)n * CC + c) = pl.u;
}

// ---------------- embedding MLP: silu(emb) @ emb_w + emb_b -> [B][512] ----------------
__global__ __launch_bounds__(256) void emb_mlp_kernel(
    const float* __restrict__ emb, const float* __restrict__ ew,
    const float* __restrict__ eb, float* __restrict__ outp)
{
    int gid = blockIdx.x * blockDim.x + threadIdx.x;
    if (gid >= B_ * 2 * CC) return;
    int b = gid >> 9;
    int j = gid & 511;
    float acc = eb[j];
    for (int i = 0; i < CE; ++i) {
        float e = emb[b * CE + i];
        acc += silu_f(e) * ew[(size_t)i * 2 * CC + j];
    }
    outp[gid] = acc;
}

// ---------------- fragment load helpers (two 16B LDS chunks -> v16bf) ----------------
union V16 { v16bf v; uint4 q[2]; };
__device__ __forceinline__ v16bf ldfrag(const __bf16* p0, const __bf16* p1) {
    V16 t;
    t.q[0] = *(const uint4*)p0;
    t.q[1] = *(const uint4*)p1;
    return t.v;
}

// ---------------- async-stage one tap's tiles into an LDS buffer ----------------
__device__ __forceinline__ void stage_tap(
    int k, int tid, int row0, int co0,
    const __bf16* __restrict__ xhi, const __bf16* __restrict__ xlo,
    const int* __restrict__ nbr,
    const __bf16* __restrict__ whi, const __bf16* __restrict__ wlo,
    __bf16* Ahi, __bf16* Alo, __bf16* Bhi, __bf16* Blo)
{
    // weights: [k][co][ci] bf16, contiguous rows of 256 -> 32 x b128 per row
    const __bf16* gh = whi + ((size_t)k * CC + co0) * CC;
    const __bf16* gl = wlo + ((size_t)k * CC + co0) * CC;
    for (int j = tid; j < COB * 32; j += 256) {
        int row = j >> 5;
        int c   = (j & 31) * 8;
        async_cp16(Bhi + row * SP + c, gh + (size_t)row * CC + c);
        async_cp16(Blo + row * SP + c, gl + (size_t)row * CC + c);
    }
    // gathered activations: 4 threads per row, 64 bf16 (8 x b128) each
    int r = tid >> 2;
    int q = (tid & 3) * 64;
    int idx = nbr[(size_t)(row0 + r) * KT + k];
    __bf16* dh = Ahi + r * SP + q;
    __bf16* dl = Alo + r * SP + q;
    if (idx < 0) {
        uint4 z = make_uint4(0u, 0u, 0u, 0u);
#pragma unroll
        for (int j = 0; j < 8; ++j) { ((uint4*)dh)[j] = z; ((uint4*)dl)[j] = z; }
    } else {
        const __bf16* shp = xhi + (size_t)idx * CC + q;
        const __bf16* slp = xlo + (size_t)idx * CC + q;
#pragma unroll
        for (int j = 0; j < 8; ++j) {
            async_cp16(dh + j * 8, shp + j * 8);
            async_cp16(dl + j * 8, slp + j * 8);
        }
    }
}

// ---------------- sparse conv: 27 gathered GEMMs, bf16x3 split WMMA, double-buffered async ----------------
__global__ __launch_bounds__(256) void sparse_conv_kernel(
    const __bf16* __restrict__ xhi, const __bf16* __restrict__ xlo,
    const int* __restrict__ nbr,
    const __bf16* __restrict__ whi, const __bf16* __restrict__ wlo,
    const float* __restrict__ bias, const float* __restrict__ resid,
    float* __restrict__ out)
{
    extern __shared__ __bf16 smem[];
    constexpr int APB = ROWS * SP;                 // elements per A plane
    constexpr int BPB = COB * SP;                  // elements per B plane
    constexpr int BUF = 2 * APB + 2 * BPB;         // elements per double-buffer slot

    const int tid  = threadIdx.x;
    const int lane = tid & 31;
    const int wave = tid >> 5;          // 8 waves = 4 co strips x 2 row halves
    const int l16  = lane & 15;
    const int h    = lane >> 4;
    const int cs   = wave & 3;          // co strip
    const int rh   = wave >> 2;         // row half
    const int row0 = blockIdx.x * ROWS;
    const int co0  = blockIdx.y * COB;

    v8f acc[2] = {};                    // two 16x16 f32 tiles per wave

    // prologue: stage tap 0 into buffer 0
    stage_tap(0, tid, row0, co0, xhi, xlo, nbr, whi, wlo,
              smem, smem + APB, smem + 2 * APB, smem + 2 * APB + BPB);
    wait_async0();
    __syncthreads();

    for (int k = 0; k < KT; ++k) {
        const int cur = k & 1;
        __bf16* base  = smem + cur * BUF;
        __bf16* nbase = smem + (1 - cur) * BUF;

        // issue next tap's DMA before this tap's math (overlaps with WMMA)
        if (k + 1 < KT) {
            stage_tap(k + 1, tid, row0, co0, xhi, xlo, nbr, whi, wlo,
                      nbase, nbase + APB, nbase + 2 * APB, nbase + 2 * APB + BPB);
        }

        // fragment base pointers for this wave
        const __bf16* Ah0 = base +             (rh * 32 + l16) * SP + 8 * h;
        const __bf16* Al0 = base + APB +       (rh * 32 + l16) * SP + 8 * h;
        const __bf16* Bh0 = base + 2 * APB +   (cs * 16 + l16) * SP + 8 * h;
        const __bf16* Bl0 = base + 2 * APB + BPB + (cs * 16 + l16) * SP + 8 * h;

        for (int kk = 0; kk < CC; kk += 32) {
            v16bf bh  = ldfrag(Bh0 + kk, Bh0 + kk + 16);
            v16bf bl  = ldfrag(Bl0 + kk, Bl0 + kk + 16);
            v16bf ah0 = ldfrag(Ah0 + kk, Ah0 + kk + 16);
            v16bf ah1 = ldfrag(Ah0 + 16 * SP + kk, Ah0 + 16 * SP + kk + 16);
            v16bf al0 = ldfrag(Al0 + kk, Al0 + kk + 16);
            v16bf al1 = ldfrag(Al0 + 16 * SP + kk, Al0 + 16 * SP + kk + 16);
            // hi*hi + hi*lo + lo*hi  (interleaved across the two acc tiles)
            acc[0] = __builtin_amdgcn_wmma_f32_16x16x32_bf16(false, ah0, false, bh, (short)0, acc[0], false, false);
            acc[1] = __builtin_amdgcn_wmma_f32_16x16x32_bf16(false, ah1, false, bh, (short)0, acc[1], false, false);
            acc[0] = __builtin_amdgcn_wmma_f32_16x16x32_bf16(false, ah0, false, bl, (short)0, acc[0], false, false);
            acc[1] = __builtin_amdgcn_wmma_f32_16x16x32_bf16(false, ah1, false, bl, (short)0, acc[1], false, false);
            acc[0] = __builtin_amdgcn_wmma_f32_16x16x32_bf16(false, al0, false, bh, (short)0, acc[0], false, false);
            acc[1] = __builtin_amdgcn_wmma_f32_16x16x32_bf16(false, al1, false, bh, (short)0, acc[1], false, false);
        }

        // all DMA for tap k+1 landed + everyone done reading buffer `cur`
        wait_async0();
        __syncthreads();
    }

    // epilogue: D layout -> lane col = l16, rows v + 8*h per VGPR v
    const int col = co0 + cs * 16 + l16;
    const float bv = bias[col];
#pragma unroll
    for (int mt = 0; mt < 2; ++mt) {
#pragma unroll
        for (int v = 0; v < 8; ++v) {
            int row = row0 + rh * 32 + mt * 16 + v + 8 * h;
            float val = acc[mt][v] + bv;
            if (resid) val += resid[(size_t)row * CC + col];
            out[(size_t)row * CC + col] = val;
        }
    }
}

// ---------------- host launcher ----------------
extern "C" void kernel_launch(void* const* d_in, const int* in_sizes, int n_in,
                              void* d_out, int out_size, void* d_ws, size_t ws_size,
                              hipStream_t stream) {
    const float* feats = (const float*)d_in[0];
    const float* emb   = (const float*)d_in[1];
    const int*   jidx  = (const int*)  d_in[2];
    const int*   nbr   = (const int*)  d_in[3];
    const float* gn1_w = (const float*)d_in[4];
    const float* gn1_b = (const float*)d_in[5];
    const float* w1    = (const float*)d_in[6];
    const float* b1    = (const float*)d_in[7];
    const float* emb_w = (const float*)d_in[8];
    const float* emb_b = (const float*)d_in[9];
    const float* gn2_w = (const float*)d_in[10];
    const float* gn2_b = (const float*)d_in[11];
    const float* w2    = (const float*)d_in[12];
    const float* b2    = (const float*)d_in[13];
    float* out = (float*)d_out;

    float* ws = (float*)d_ws;
    const size_t NC  = (size_t)N_ * CC;      // 51.2M
    const size_t WKF = (size_t)WK / 2;       // weight bf16 buffer in float units
    __bf16* xhi  = (__bf16*)ws;              // NC bf16 = NC/2 floats
    __bf16* xlo  = (__bf16*)(ws + NC / 2);
    float*  hbuf = ws + NC;                  // conv1 output, fp32
    __bf16* w1hi = (__bf16*)(ws + 2 * NC);
    __bf16* w1lo = (__bf16*)(ws + 2 * NC + WKF);
    __bf16* w2hi = (__bf16*)(ws + 2 * NC + 2 * WKF);
    __bf16* w2lo = (__bf16*)(ws + 2 * NC + 3 * WKF);
    float* st1  = ws + 2 * NC + 4 * WKF;     // 128
    float* sq1  = st1 + 128;
    float* cn1  = sq1 + 128;                 // 4
    float* st2  = cn1 + 4;
    float* sq2  = st2 + 128;
    float* cn2  = sq2 + 128;
    float* embo = cn2 + 4;                   // 2048

    constexpr size_t shmem = (size_t)2 * (2 * ROWS * SP + 2 * COB * SP) * sizeof(__bf16); // 270,336 B
    hipFuncSetAttribute((const void*)sparse_conv_kernel,
                        hipFuncAttributeMaxDynamicSharedMemorySize, (int)shmem);

    const int statsN = 128 + 128 + 4 + 128 + 128 + 4 + 2048;
    zero_kernel<<<(statsN + 255) / 256, 256, 0, stream>>>(st1, statsN);

    // split + transpose both weight tensors to bf16 hi/lo [k][co][ci]
    wsplit_kernel<<<(WK + 255) / 256, 256, 0, stream>>>(w1, w1hi, w1lo);
    wsplit_kernel<<<(WK + 255) / 256, 256, 0, stream>>>(w2, w2hi, w2lo);

    // GN1 -> SiLU -> bf16 hi/lo
    {
        long long t = (long long)N_ * G_;
        gn_stats_kernel<<<(unsigned)((t + 255) / 256), 256, 0, stream>>>(feats, jidx, st1, sq1, cn1);
        long long t2 = (long long)N_ * (CC / 4);
        gn_apply_kernel<<<(unsigned)((t2 + 255) / 256), 256, 0, stream>>>(
            feats, jidx, st1, sq1, cn1, gn1_w, gn1_b, nullptr, xhi, xlo);
    }
    // conv1 -> hbuf
    dim3 cgrid(N_ / ROWS, CC / COB);
    sparse_conv_kernel<<<cgrid, 256, shmem, stream>>>(xhi, xlo, nbr, w1hi, w1lo, b1, nullptr, hbuf);

    // embedding MLP -> embo (scale||shift)
    emb_mlp_kernel<<<(B_ * 2 * CC + 255) / 256, 256, 0, stream>>>(emb, emb_w, emb_b, embo);

    // GN2 + FiLM -> SiLU -> bf16 hi/lo
    {
        long long t = (long long)N_ * G_;
        gn_stats_kernel<<<(unsigned)((t + 255) / 256), 256, 0, stream>>>(hbuf, jidx, st2, sq2, cn2);
        long long t2 = (long long)N_ * (CC / 4);
        gn_apply_kernel<<<(unsigned)((t2 + 255) / 256), 256, 0, stream>>>(
            hbuf, jidx, st2, sq2, cn2, gn2_w, gn2_b, embo, xhi, xlo);
    }
    // conv2 + b2 + residual -> out
    sparse_conv_kernel<<<cgrid, 256, shmem, stream>>>(xhi, xlo, nbr, w2hi, w2lo, b2, feats, out);
}
